// ConvolutionalCapsule_5488968204915
// MI455X (gfx1250) — matrix operand order repaired and checked by reading, not compiled, not run on hardware
//
#include <hip/hip_runtime.h>

// CDNA5 (gfx1250) capsule-conv + dynamic routing.
//  prep : f32 -> bf16 weight copy + zero-halo-padded bf16 input (enables async DMA)
//  conv : implicit-GEMM via v_wmma_f32_16x16x32_bf16 (M=540800,K=256,N=128),
//         double-buffered LDS pipeline fed ONLY by global_load_async_to_lds_*
//         (ASYNCcnt), fragment-native LDS layouts -> 1 vector LDS load/fragment.
//  route: per-pixel fused dynamic routing, all 3 iterations LDS-resident.

typedef __attribute__((ext_vector_type(16))) __bf16 v16bf;
typedef __attribute__((ext_vector_type(8)))  __bf16 v8bf;
typedef __attribute__((ext_vector_type(8)))  float  v8f;

#define OW      65
#define PIX     (OW * OW)        // 4225
#define NCH     128              // OUT_CAPS*OUT_CH
#define KTOT    256              // Cin*4*4
#define MTILE   128              // rows per workgroup (540800 / 128 = 4225)
#define KSTEP   32               // K per WMMA
#define XP      132              // padded spatial extent (2-halo)
#define XPP     (XP * XP)        // 17424

// ---- CDNA5 async global->LDS copy (ISA 10.?, tracked by ASYNCcnt) ----------
__device__ __forceinline__ void async_b32(unsigned lds, const void* g) {
  asm volatile("global_load_async_to_lds_b32 %0, %1, off"
               :: "v"(lds), "v"(g) : "memory");
}
__device__ __forceinline__ void async_b128(unsigned lds, const void* g) {
  asm volatile("global_load_async_to_lds_b128 %0, %1, off"
               :: "v"(lds), "v"(g) : "memory");
}
__device__ __forceinline__ void wait_async0() {
  asm volatile("s_wait_asynccnt 0x0" ::: "memory");
}
__device__ __forceinline__ unsigned lds_addr(const void* p) {
  return (unsigned)(unsigned long long)p;   // LDS address = low 32 bits of the pointer
}

// ---------------------------------------------------------------------------
// prep: weights f32 -> bf16
// ---------------------------------------------------------------------------
__global__ __launch_bounds__(256) void prep_w(const float* __restrict__ Wc,
                                              __bf16* __restrict__ Wb) {
  const int i = blockIdx.x * 256 + threadIdx.x;   // 32768 elems
  Wb[i] = (__bf16)Wc[i];
}

// ---------------------------------------------------------------------------
// prep: input f32 [128][16][128][128] -> bf16 zero-padded [128][16][132][132]
// ---------------------------------------------------------------------------
__global__ __launch_bounds__(256) void prep_x(const float* __restrict__ x,
                                              __bf16* __restrict__ xb) {
  const size_t idx = (size_t)blockIdx.x * 256 + threadIdx.x;
  if (idx >= (size_t)2048 * XPP) return;
  const int nc   = (int)(idx / XPP);      // n*16 + ci
  const int rem  = (int)(idx % XPP);
  const int ixp  = rem / XP;
  const int iyp  = rem % XP;
  const int ix   = ixp - 2;
  const int iy   = iyp - 2;
  float v = 0.0f;
  if ((unsigned)ix < 128u && (unsigned)iy < 128u)
    v = x[(size_t)nc * 16384 + ix * 128 + iy];
  xb[idx] = (__bf16)v;
}

// ---------------------------------------------------------------------------
// Conv as implicit GEMM:  u_hat[r][c] = sum_k patch(r,k) * W[c][k] + bias[c]
//   r = (n, ox, oy) ; k = ci*16 + kx*4 + ky ; c = oc*16 + cout
// 256 threads = 8 waves. Tile M=128, N=128. Wave w: M-strip w, all 8 N-tiles
// -> 8 accumulators, 8 WMMAs per k-step per wave.
// A-tile LDS layout uses the K bit-3<->4 swap so lane-half hi reads its 16
// K-values (ISA 16-bit A 16x32 order) as one contiguous 32-byte run.
// ---------------------------------------------------------------------------
__global__ __launch_bounds__(256) void caps_conv_wmma(
    const __bf16* __restrict__ xb,      // [128][16][132][132] padded
    const __bf16* __restrict__ Wb,      // [128][256]
    const float*  __restrict__ bconv,   // [128]
    __bf16* __restrict__ uhat)          // [540800][128] bf16
{
  __shared__ __align__(32) __bf16 Ald[2][MTILE * 32];  // 2 x 8 KB, perm-K
  __shared__ __align__(32) __bf16 Bld[2][NCH * 32];    // 2 x 8 KB, [col][k]

  const int tid  = threadIdx.x;
  const int lane = tid & 31;
  const int wave = tid >> 5;            // M-strip
  const int lr   = lane & 15;
  const int hi   = lane >> 4;

  const int mbase = blockIdx.x * MTILE;

  // ---- A async-staging assignment: row = tid&127, ci-half = tid>>7 ----
  const int arow   = tid & 127;
  const int cihalf = tid >> 7;          // k-halves 0..15 / 16..31 == ci parity
  const int pos0   = cihalf << 3;       // perm(akq0): 0 or 8
  const int r    = mbase + arow;
  const int n    = r / PIX;
  const int rem  = r - n * PIX;
  const int ox   = rem / OW;
  const int oy   = rem - ox * OW;
  // padded-row base for this output pixel: halo makes all reads in-bounds
  const __bf16* xrow = xb + (size_t)(n * 16) * XPP + (2 * ox) * XP + 2 * oy;
  const unsigned aldsb[2] = { lds_addr(&Ald[0][arow * 32]),
                              lds_addr(&Ald[1][arow * 32]) };

  // ---- B async-staging assignment: col = tid>>1, k-half = (tid&1)*16 ----
  const int bcol = tid >> 1;
  const int bkq0 = (tid & 1) << 4;
  const __bf16* wrow = Wb + bcol * KTOT + bkq0;
  const unsigned bldsb[2] = { lds_addr(&Bld[0][bcol * 32 + ((tid & 1) << 3)]),
                              lds_addr(&Bld[1][bcol * 32 + ((tid & 1) << 3)]) };

  v8f acc[8] = {};

  // stage(ks, buf): pure async DMA, no VGPR data movement
  auto stage = [&](int ks, int buf) {
    const __bf16* ga = xrow + (size_t)(2 * ks + cihalf) * XPP;  // this ci
    #pragma unroll
    for (int kx = 0; kx < 4; ++kx) {
      // 4 ky-elems = 8 bytes, global byte addr = multiple of 4 (starts 4*oy)
      const __bf16* g = ga + kx * XP;
      const unsigned l =
          aldsb[buf] + (unsigned)((pos0 + (kx & 1) * 4 + (kx >> 1) * 16) * 2);
      async_b32(l,     g);
      async_b32(l + 4, g + 2);
    }
    const __bf16* gw = wrow + ks * KSTEP;      // 16 elems, 32B, 32B-aligned
    async_b128(bldsb[buf],      gw);
    async_b128(bldsb[buf] + 32, gw + 8);
  };

  stage(0, 0);
  wait_async0();
  __syncthreads();

  for (int ks = 0; ks < KTOT / KSTEP; ++ks) {
    const int cur = ks & 1;
    if (ks < KTOT / KSTEP - 1)
      stage(ks + 1, cur ^ 1);                  // DMA next step into other buf

    // A fragment: single contiguous 32-byte LDS load (perm layout)
    const v16bf afrag =
        *(const v16bf*)&Ald[cur][(wave * 16 + lr) * 32 + hi * 16];

    #pragma unroll
    for (int t = 0; t < 8; ++t) {
      const v16bf bfrag =
          *(const v16bf*)&Bld[cur][(t * 16 + lr) * 32 + hi * 16];
      acc[t] = __builtin_amdgcn_wmma_f32_16x16x32_bf16(
          false, afrag, false, bfrag, (short)0, acc[t], false, false);
    }
    wait_async0();                             // DMA for next buf complete
    __syncthreads();
  }

  // Epilogue: C/D layout (VGPR i, half hi -> M = i + 8*hi; N = lr)
  #pragma unroll
  for (int t = 0; t < 8; ++t) {
    const int col    = t * 16 + lr;
    const float bias = bconv[col];
    #pragma unroll
    for (int i = 0; i < 8; ++i) {
      const int m = wave * 16 + i + 8 * hi;
      uhat[(size_t)(mbase + m) * NCH + col] = (__bf16)(acc[t][i] + bias);
    }
  }
}

// ---------------------------------------------------------------------------
// Fused dynamic routing: one workgroup per spatial pixel.
// u_hat[pixel] = 128x128 (bf16 in HBM) -> 64 KB f32 in LDS for all 3 iters.
// ---------------------------------------------------------------------------
__global__ __launch_bounds__(256) void caps_route(
    const __bf16* __restrict__ uhat,    // [540800][128] = [n][ox][oy][c] bf16
    float* __restrict__ out)            // [16][8][16][65][65]
{
  __shared__ float u[128 * NCH];        // 64 KB: [n=b*8+i][c=oc*16+cout]
  __shared__ float vj[2048];            // [b][oc][cout]
  __shared__ float bij[64];             // [i][oc]
  __shared__ float cij[64];             // [i][oc]
  __shared__ float scale[128];          // [b][oc]

  const int tid = threadIdx.x;
  const int pix = blockIdx.x;           // ox*65 + oy

  // Load slab: 8 bf16 (16B vector) per thread-iteration, convert to f32.
  for (int e = tid * 8; e < 128 * NCH; e += 256 * 8) {
    const int nn = e >> 7;
    const int c  = e & 127;
    const v8bf vv =
        *(const v8bf*)&uhat[(size_t)nn * PIX * NCH + (size_t)pix * NCH + c];
    #pragma unroll
    for (int j = 0; j < 8; ++j) u[e + j] = (float)vv[j];
  }
  if (tid < 64) bij[tid] = 0.0f;
  __syncthreads();

  for (int it = 0; it < 3; ++it) {
    // c_ij = softmax over input capsules i (per oc)
    if (tid < 8) {
      const int oc = tid;
      float mx = -1e30f;
      for (int i = 0; i < 8; ++i) mx = fmaxf(mx, bij[i * 8 + oc]);
      float ex[8];
      float s = 0.0f;
      for (int i = 0; i < 8; ++i) { ex[i] = __expf(bij[i * 8 + oc] - mx); s += ex[i]; }
      const float inv = 1.0f / s;
      for (int i = 0; i < 8; ++i) cij[i * 8 + oc] = ex[i] * inv;
    }
    __syncthreads();

    // s_j[b][oc][cout] = sum_i c_ij[i][oc] * u[b][i][oc][cout]
    for (int e = tid; e < 2048; e += 256) {
      const int b  = e >> 7;
      const int rm = e & 127;           // oc*16 + cout
      const int oc = rm >> 4;
      float s = 0.0f;
      #pragma unroll
      for (int i = 0; i < 8; ++i)
        s += cij[i * 8 + oc] * u[(b * 8 + i) * NCH + rm];
      vj[e] = s;
    }
    __syncthreads();

    // squash scale per (b, oc): sn / ((1+sn)*sqrt(sn))
    if (tid < 128) {
      const float* vp = &vj[tid * 16];
      float sn = 0.0f;
      #pragma unroll
      for (int c = 0; c < 16; ++c) sn += vp[c] * vp[c];
      scale[tid] = sn / ((1.0f + sn) * sqrtf(sn));
    }
    __syncthreads();
    for (int e = tid; e < 2048; e += 256) vj[e] *= scale[e >> 4];
    __syncthreads();

    // b_ij += mean_b  dot_cout( u[b][i][oc][:], v[b][oc][:] )
    if (it < 2) {
      if (tid < 64) {
        const int i  = tid >> 3;
        const int oc = tid & 7;
        float acc = 0.0f;
        for (int b = 0; b < 16; ++b) {
          const float* up = &u[(b * 8 + i) * NCH + oc * 16];
          const float* vp = &vj[(b * 8 + oc) * 16];
          #pragma unroll
          for (int c = 0; c < 16; ++c) acc += up[c] * vp[c];
        }
        bij[tid] += acc * (1.0f / 16.0f);
      }
      __syncthreads();
    }
  }

  // out[b][oc][cout][ox][oy]
  for (int e = tid; e < 2048; e += 256)
    out[(size_t)e * PIX + pix] = vj[e];
}

extern "C" void kernel_launch(void* const* d_in, const int* in_sizes, int n_in,
                              void* d_out, int out_size, void* d_ws, size_t ws_size,
                              hipStream_t stream) {
  (void)in_sizes; (void)n_in; (void)out_size; (void)ws_size;
  const float* x     = (const float*)d_in[0];   // (16,8,16,128,128)
  const float* Wconv = (const float*)d_in[1];   // (128,16,4,4)
  const float* bconv = (const float*)d_in[2];   // (128,)
  float* out = (float*)d_out;                   // (16,8,16,65,65)

  // Workspace carve-up (all 256B aligned):
  //   Wb   : 32768 bf16            =     64 KB
  //   xb   : 2048*132*132 bf16     = 71,368,704 B
  //   uhat : 540800*128 bf16       = 138,444,800 B     (total ~200 MB)
  char* ws = (char*)d_ws;
  __bf16* Wb   = (__bf16*)ws;
  __bf16* xb   = (__bf16*)(ws + 65536);
  __bf16* uhat = (__bf16*)(ws + 65536 + 71368704);

  prep_w<<<32768 / 256, 256, 0, stream>>>(Wconv, Wb);
  prep_x<<<(int)(((size_t)2048 * XPP + 255) / 256), 256, 0, stream>>>(x, xb);
  caps_conv_wmma<<<(128 * PIX) / MTILE, 256, 0, stream>>>(xb, Wb, bconv, uhat);
  caps_route<<<PIX, 256, 0, stream>>>(uhat, out);
}
